// Decoder_20486994002402
// MI455X (gfx1250) — matrix operand order repaired and checked by reading, not compiled, not run on hardware
//
#include <hip/hip_runtime.h>
#include <hip/hip_bf16.h>
#include <math.h>

// ---------------- problem constants ----------------
constexpr int B    = 128;
constexpr int S    = 512;
constexpr int H    = 1024;
constexpr int E    = 100;
constexpr int V    = 50000;
constexpr int Q    = E + 2 * H;      // 2148 : attention query width
constexpr int XDIM = E + H;          // 1124 : lstm0 input width
constexpr int VS   = V + S;          // 50512 : extended vocab

typedef __attribute__((ext_vector_type(16))) __bf16 v16bf;
typedef __attribute__((ext_vector_type(8)))  float  v8f;
typedef __attribute__((ext_vector_type(4)))  float  f4;   // native clang vector
                                                          // (HIP float4 is a class
                                                          //  -> rejected by
                                                          //  __builtin_nontemporal_load)

// CDNA5 async global->LDS path. Builtin parameters (confirmed by diagnostics):
//   (v4i addrspace(1)*, v4i addrspace(3)*, imm offset, imm cpol)
#if defined(__HIP_DEVICE_COMPILE__) && __has_builtin(__builtin_amdgcn_global_load_async_to_lds_b128)
#define USE_ASYNC_LDS 1
typedef int v4i_gcc __attribute__((vector_size(16)));
typedef __attribute__((address_space(1))) v4i_gcc GV4;   // global
typedef __attribute__((address_space(3))) v4i_gcc LV4;   // LDS
#endif

// ---------------------------------------------------------------------------
// bf16-WMMA GEMM:  C[M x N] = A[M x K] * B[N x K]^T (+bias0+bias1)(+C)
// A, B fp32 row-major, converted to bf16 fragments in-register.
// One wave computes a 16x32 C tile: one A fragment feeds two independent
// v_wmma_f32_16x16x32_bf16 accumulator chains per 32-wide K step.
// Raw loads are batched ahead of all conversions so the backend can emit a
// single load clause per K step; B (weights, streamed once, > L2) uses
// non-temporal loads, A (tiny activations, reused by every N tile) stays RT.
//
// Fragment layouts used here (wave32, same contiguous k order for A and B):
//   A 16x32 bf16: lane L holds row m = L%16; element j -> k = 16*(L/16)+j
//   B 32x16 bf16: lane L holds col n = L%16; element j -> k = 16*(L/16)+j
//   C/D f32:      element r -> m = r + 8*(L/16), n = L%16
// ---------------------------------------------------------------------------
__device__ __forceinline__ void cvt4(v16bf& d, int off, f4 x) {
    d[off + 0] = (__bf16)x.x; d[off + 1] = (__bf16)x.y;
    d[off + 2] = (__bf16)x.z; d[off + 3] = (__bf16)x.w;
}

__global__ __launch_bounds__(128) void gemm_bf16_wmma(
    const float* __restrict__ A, int lda,
    const float* __restrict__ Bw, int ldb,
    float* __restrict__ C, int ldc,
    int M, int N, int K, int accumulate,
    const float* __restrict__ bias0, const float* __restrict__ bias1)
{
    const int wave   = threadIdx.x >> 5;
    const int lane   = threadIdx.x & 31;
    const int tilesN = N >> 4;
    const int pairsN = (tilesN + 1) >> 1;
    const int tilesM = (M + 15) >> 4;
    const int t = blockIdx.x * 4 + wave;
    if (t >= tilesM * pairsN) return;          // wave-uniform: EXEC stays full
    const int m0   = (t / pairsN) << 4;
    const int n0   = (t % pairsN) << 5;        // 32-wide N tile
    const int half = lane >> 4;                // 0 or 1
    const int sub  = lane & 15;

    const int n0s = n0 + sub;                  // always < N
    const int n1s = n0 + 16 + sub;             // may exceed N when tilesN odd
    const int n1c = (n1s < N) ? n1s : (N - 1); // clamp for safe loads

    const float* arow  = A  + (size_t)(m0 + sub) * lda;
    const float* brow0 = Bw + (size_t)n0s * ldb;
    const float* brow1 = Bw + (size_t)n1c * ldb;

    v8f acc0 = {0.f, 0.f, 0.f, 0.f, 0.f, 0.f, 0.f, 0.f};
    v8f acc1 = {0.f, 0.f, 0.f, 0.f, 0.f, 0.f, 0.f, 0.f};

    const int kmain = K & ~31;
    #pragma unroll 2
    for (int k0 = 0; k0 < kmain; k0 += 32) {   // branch-free main loop
        const f4* pa0 = (const f4*)(arow  + k0 + 16 * half);
        const f4* pa1 = (const f4*)(arow  + k0 + 16 * half + 8);
        const f4* pb0 = (const f4*)(brow0 + k0 + 16 * half);
        const f4* pb1 = (const f4*)(brow1 + k0 + 16 * half);
        // --- batch ALL raw loads first (one clause, one wait) ---
        f4 ra0 = pa0[0], ra1 = pa0[1];
        f4 ra2 = pa1[0], ra3 = pa1[1];
        f4 rb00 = __builtin_nontemporal_load(pb0 + 0);
        f4 rb01 = __builtin_nontemporal_load(pb0 + 1);
        f4 rb02 = __builtin_nontemporal_load(pb0 + 2);
        f4 rb03 = __builtin_nontemporal_load(pb0 + 3);
        f4 rb10 = __builtin_nontemporal_load(pb1 + 0);
        f4 rb11 = __builtin_nontemporal_load(pb1 + 1);
        f4 rb12 = __builtin_nontemporal_load(pb1 + 2);
        f4 rb13 = __builtin_nontemporal_load(pb1 + 3);
        // --- then convert everything ---
        v16bf a, b0, b1;
        cvt4(a,  0, ra0);  cvt4(a,  4, ra1);  cvt4(a,  8, ra2);  cvt4(a, 12, ra3);
        cvt4(b0, 0, rb00); cvt4(b0, 4, rb01); cvt4(b0, 8, rb02); cvt4(b0, 12, rb03);
        cvt4(b1, 0, rb10); cvt4(b1, 4, rb11); cvt4(b1, 8, rb12); cvt4(b1, 12, rb13);
        acc0 = __builtin_amdgcn_wmma_f32_16x16x32_bf16(
            false, a, false, b0, (short)0, acc0, false, false);
        acc1 = __builtin_amdgcn_wmma_f32_16x16x32_bf16(
            false, a, false, b1, (short)0, acc1, false, false);
    }
    if (kmain < K) {                           // single guarded K tail
        v16bf a, b0, b1;
        for (int j = 0; j < 8; ++j) {
            int ka = kmain + 16 * half + j;
            a[j]     = (__bf16)((ka      < K) ? arow[ka]      : 0.f);
            a[j + 8] = (__bf16)((ka + 8 < K) ? arow[ka + 8] : 0.f);
        }
        for (int j = 0; j < 16; ++j) {
            int kb = kmain + 16 * half + j;
            b0[j] = (__bf16)((kb < K) ? brow0[kb] : 0.f);
            b1[j] = (__bf16)((kb < K) ? brow1[kb] : 0.f);
        }
        acc0 = __builtin_amdgcn_wmma_f32_16x16x32_bf16(
            false, a, false, b0, (short)0, acc0, false, false);
        acc1 = __builtin_amdgcn_wmma_f32_16x16x32_bf16(
            false, a, false, b1, (short)0, acc1, false, false);
    }

    float badd0 = 0.f, badd1 = 0.f;
    if (bias0) { badd0 += bias0[n0s]; if (n1s < N) badd1 += bias0[n1s]; }
    if (bias1) { badd0 += bias1[n0s]; if (n1s < N) badd1 += bias1[n1s]; }
    #pragma unroll
    for (int r = 0; r < 8; ++r) {
        int m = m0 + r + 8 * half;
        if (m < M) {
            size_t i0 = (size_t)m * ldc + n0s;
            float v0 = acc0[r] + badd0;
            if (accumulate) v0 += C[i0];
            C[i0] = v0;
            if (n1s < N) {
                size_t i1 = (size_t)m * ldc + n1s;
                float v1 = acc1[r] + badd1;
                if (accumulate) v1 += C[i1];
                C[i1] = v1;
            }
        }
    }
}

// ---------------------------------------------------------------------------
// prep: embedding gather + query = [h0.reshape(B,-1) | emb]  (raw flat reshape)
// ---------------------------------------------------------------------------
__global__ void prep_kernel(const int* __restrict__ ids,
                            const float* __restrict__ h0flat,
                            const float* __restrict__ embT,
                            float* __restrict__ query,
                            float* __restrict__ emb)
{
    int idx = blockIdx.x * blockDim.x + threadIdx.x;
    if (idx >= B * Q) return;
    int b = idx / Q, j = idx % Q;
    float val;
    if (j < 2 * H) {
        val = h0flat[b * 2 * H + j];         // raw row-major reshape of (2,B,H)
    } else {
        int e = j - 2 * H;
        val = embT[(size_t)ids[b] * E + e];  // padding_idx=0 row is already 0
        emb[b * E + e] = val;
    }
    query[(size_t)b * Q + j] = val;
}

// ---------------------------------------------------------------------------
// Single-pass flash-style attention (one block / batch row, 512 threads).
// 8 encoder rows at a time are staged into LDS (async-to-LDS when available),
// scores + online softmax + context accumulate from LDS: enc is read ONCE.
// ---------------------------------------------------------------------------
constexpr int CH = 8;   // encoder rows per LDS chunk (32 KB)

__global__ __launch_bounds__(512) void attn_kernel(
    const float* __restrict__ enc, const int* __restrict__ masks,
    const float* __restrict__ v,   const float* __restrict__ emb,
    float* __restrict__ attn, float* __restrict__ rnn_in)
{
    __shared__ __align__(16) float ldsE[CH * H];  // 32 KB chunk
    __shared__ float sv[H];                       // 4 KB
    __shared__ float sc[S];                       // 2 KB raw scores
    __shared__ float csc[CH];
    const int b    = blockIdx.x;
    const int tid  = threadIdx.x;
    const int wave = tid >> 5;
    const int lane = tid & 31;

    for (int i = tid; i < H; i += 512) sv[i] = v[b * H + i];

    const float* erow = enc + (size_t)b * S * H;
    float runM = -__builtin_inff(), runZ = 0.f, ctx0 = 0.f, ctx1 = 0.f;

    for (int s0 = 0; s0 < S; s0 += CH) {
        // stage CH*H contiguous floats into LDS
        const float* src = erow + (size_t)s0 * H;
        for (int i = tid; i < CH * H / 4; i += 512) {
#if USE_ASYNC_LDS
            __builtin_amdgcn_global_load_async_to_lds_b128(
                (GV4*)(unsigned long long)(uintptr_t)(src + i * 4),
                (LV4*)(unsigned)(uintptr_t)(&ldsE[i * 4]), 0, 0);
#else
            ((f4*)ldsE)[i] = ((const f4*)src)[i];
#endif
        }
#if USE_ASYNC_LDS
#if __has_builtin(__builtin_amdgcn_s_wait_asynccnt)
        __builtin_amdgcn_s_wait_asynccnt(0);
#endif
#endif
        __syncthreads();

        // scores for the chunk: waves 0..CH-1, one cooperative dot each
        if (wave < CH) {
            const float* ep = ldsE + wave * H;
            float acc = 0.f;
            for (int hh = lane; hh < H; hh += 32) acc += ep[hh] * sv[hh];
            for (int off = 16; off > 0; off >>= 1) acc += __shfl_xor(acc, off, 32);
            if (lane == 0) {
                float sval = masks[b * S + s0 + wave] ? -__builtin_inff() : acc;
                csc[wave] = sval;
                sc[s0 + wave] = sval;
            }
        }
        __syncthreads();

        // online softmax update (redundant per-thread, deterministic)
        float cmax = -__builtin_inff();
        #pragma unroll
        for (int s = 0; s < CH; ++s) cmax = fmaxf(cmax, csc[s]);
        float newM  = fmaxf(runM, cmax);
        float scale = __expf(runM - newM);
        float w[CH], zadd = 0.f;
        #pragma unroll
        for (int s = 0; s < CH; ++s) { w[s] = __expf(csc[s] - newM); zadd += w[s]; }
        runZ = runZ * scale + zadd;
        float a0 = ctx0 * scale, a1 = ctx1 * scale;
        #pragma unroll
        for (int s = 0; s < CH; ++s) {
            a0 += w[s] * ldsE[s * H + tid];
            a1 += w[s] * ldsE[s * H + tid + 512];
        }
        ctx0 = a0; ctx1 = a1; runM = newM;
        __syncthreads();   // ldsE reuse fence
    }

    // finalize: normalized attention + context -> rnn_in[:, E:], emb -> [:E]
    float aw = __expf(sc[tid] - runM) / runZ;
    attn[b * S + tid] = aw;
    rnn_in[(size_t)b * XDIM + E + tid]       = ctx0 / runZ;
    rnn_in[(size_t)b * XDIM + E + tid + 512] = ctx1 / runZ;
    if (tid < E) rnn_in[(size_t)b * XDIM + tid] = emb[b * E + tid];
}

// ---------------------------------------------------------------------------
// LSTM cell pointwise (PyTorch gate order i,f,g,o)
// ---------------------------------------------------------------------------
__global__ void lstm_cell_kernel(const float* __restrict__ g,
                                 const float* __restrict__ cprev,
                                 float* __restrict__ hout,
                                 float* __restrict__ cout)
{
    int idx = blockIdx.x * blockDim.x + threadIdx.x;
    if (idx >= B * H) return;
    int b = idx / H, hh = idx % H;
    const float* gr = g + (size_t)b * 4 * H;
    float gi = gr[hh], gf = gr[H + hh], gg = gr[2 * H + hh], go = gr[3 * H + hh];
    float si = 1.f / (1.f + __expf(-gi));
    float sf = 1.f / (1.f + __expf(-gf));
    float so = 1.f / (1.f + __expf(-go));
    float c2 = sf * cprev[idx] + si * tanhf(gg);
    hout[idx] = so * tanhf(c2);
    cout[idx] = c2;
}

// ---------------------------------------------------------------------------
// p_gen = sigmoid([h2 | ctx | emb] . W_pg + b_pg), one wave per batch row
// ---------------------------------------------------------------------------
__global__ __launch_bounds__(128) void pgen_kernel(
    const float* __restrict__ h2, const float* __restrict__ rnn_in,
    const float* __restrict__ emb, const float* __restrict__ W_pg,
    const float* __restrict__ b_pg, float* __restrict__ pgen)
{
    int wave = threadIdx.x >> 5, lane = threadIdx.x & 31;
    int b = blockIdx.x * 4 + wave;
    if (b >= B) return;
    float acc = 0.f;
    for (int i = lane; i < Q; i += 32) {
        float xv;
        if (i < H)          xv = h2[b * H + i];
        else if (i < 2 * H) xv = rnn_in[(size_t)b * XDIM + E + (i - H)];
        else                xv = emb[b * E + (i - 2 * H)];
        acc += xv * W_pg[i];
    }
    for (int off = 16; off > 0; off >>= 1) acc += __shfl_xor(acc, off, 32);
    if (lane == 0) pgen[b] = 1.f / (1.f + __expf(-(acc + b_pg[0])));
}

// ---------------------------------------------------------------------------
// per-row: softmax(logits) * p_gen + 1e-10, tail [V,V+S) = 1e-10 (in d_out)
// ---------------------------------------------------------------------------
__global__ __launch_bounds__(1024) void softmax_out_kernel(
    float* __restrict__ finalp, const float* __restrict__ pgen)
{
    __shared__ float red[1024];
    const int b = blockIdx.x, tid = threadIdx.x;
    float* row = finalp + (size_t)b * VS;

    float mx = -__builtin_inff();
    for (int j = tid; j < V; j += 1024) mx = fmaxf(mx, row[j]);
    red[tid] = mx; __syncthreads();
    for (int st = 512; st > 0; st >>= 1) {
        if (tid < st) red[tid] = fmaxf(red[tid], red[tid + st]);
        __syncthreads();
    }
    mx = red[0]; __syncthreads();

    float sum = 0.f;
    for (int j = tid; j < V; j += 1024) sum += __expf(row[j] - mx);
    red[tid] = sum; __syncthreads();
    for (int st = 512; st > 0; st >>= 1) {
        if (tid < st) red[tid] += red[tid + st];
        __syncthreads();
    }
    float scale = pgen[b] / red[0];

    for (int j = tid; j < V; j += 1024) row[j] = __expf(row[j] - mx) * scale + 1e-10f;
    for (int j = V + tid; j < VS; j += 1024) row[j] = 1e-10f;
}

// ---------------------------------------------------------------------------
// scatter-add of (1 - p_gen) * attn at extend_vocab indices, then log
// ---------------------------------------------------------------------------
__global__ void scatter_kernel(float* __restrict__ finalp,
                               const int* __restrict__ ext,
                               const float* __restrict__ attn,
                               const float* __restrict__ pgen)
{
    int idx = blockIdx.x * blockDim.x + threadIdx.x;
    if (idx >= B * S) return;
    int b = idx / S;
    float add = (1.f - pgen[b]) * attn[idx];
    atomicAdd(finalp + (size_t)b * VS + ext[idx], add);
}

__global__ void log_kernel(float* __restrict__ finalp, int n)
{
    int idx = blockIdx.x * blockDim.x + threadIdx.x;
    if (idx >= n) return;
    finalp[idx] = __logf(finalp[idx] + 1e-7f);
}

// ---------------------------------------------------------------------------
extern "C" void kernel_launch(void* const* d_in, const int* in_sizes, int n_in,
                              void* d_out, int out_size, void* d_ws, size_t ws_size,
                              hipStream_t stream)
{
    const int*   input_ids = (const int*)  d_in[0];
    const int*   masks     = (const int*)  d_in[1];
    const float* h0        = (const float*)d_in[2];   // (2,B,H) flat
    const float* c0        = (const float*)d_in[3];
    const float* enc       = (const float*)d_in[4];   // (B,S,H)
    const int*   extv      = (const int*)  d_in[5];   // (B,S)
    const float* embT      = (const float*)d_in[6];   // (V+MAXLEN, E)
    const float* attn_W    = (const float*)d_in[7];   // (H, Q) row-major
    const float* W_ih0     = (const float*)d_in[8];   // (4H, E+H)
    const float* W_hh0     = (const float*)d_in[9];   // (4H, H)
    const float* b_ih0     = (const float*)d_in[10];
    const float* b_hh0     = (const float*)d_in[11];
    const float* W_ih1     = (const float*)d_in[12];  // (4H, H)
    const float* W_hh1     = (const float*)d_in[13];
    const float* b_ih1     = (const float*)d_in[14];
    const float* b_hh1     = (const float*)d_in[15];
    const float* W_out     = (const float*)d_in[16];  // (V, H)
    const float* b_out     = (const float*)d_in[17];
    const float* W_pg      = (const float*)d_in[18];  // (1, Q)
    const float* b_pg      = (const float*)d_in[19];

    // workspace carve-up (fp32)
    float* wsf    = (float*)d_ws;
    float* query  = wsf;                       // B*Q
    float* emb    = query  + (size_t)B * Q;    // B*E
    float* vbh    = emb    + (size_t)B * E;    // B*H
    float* attn   = vbh    + (size_t)B * H;    // B*S
    float* rnn_in = attn   + (size_t)B * S;    // B*XDIM
    float* gates0 = rnn_in + (size_t)B * XDIM; // B*4H
    float* gates1 = gates0 + (size_t)B * 4 * H;// B*4H
    float* pgen   = gates1 + (size_t)B * 4 * H;// B

    // d_out carve-up: final (B*VS), h_stack (2*B*H), c_stack (2*B*H)
    float* finalp = (float*)d_out;
    float* hstack = finalp + (size_t)B * VS;
    float* cstack = hstack + (size_t)2 * B * H;
    float* h1 = hstack;             float* h2 = hstack + (size_t)B * H;
    float* c1 = cstack;             float* c2 = cstack + (size_t)B * H;

    auto gemm = [&](const float* A, int lda, const float* Bw, int ldb,
                    float* C, int ldc, int M, int N, int K, int acc,
                    const float* bi0, const float* bi1) {
        int pairsN = ((N >> 4) + 1) >> 1;
        int tiles  = ((M + 15) / 16) * pairsN;
        hipLaunchKernelGGL(gemm_bf16_wmma, dim3((tiles + 3) / 4), dim3(128), 0, stream,
                           A, lda, Bw, ldb, C, ldc, M, N, K, acc, bi0, bi1);
    };

    // 1) embedding gather + query build
    hipLaunchKernelGGL(prep_kernel, dim3((B * Q + 255) / 256), dim3(256), 0, stream,
                       input_ids, h0, embT, query, emb);

    // 2) v = query @ attn_W^T   (re-associated attention: avoids the 288-GFLOP x_proj)
    gemm(query, Q, attn_W, Q, vbh, H, B, H, Q, 0, nullptr, nullptr);

    // 3) single-pass masked softmax attention + context (-> rnn_in)
    hipLaunchKernelGGL(attn_kernel, dim3(B), dim3(512), 0, stream,
                       enc, masks, vbh, emb, attn, rnn_in);

    // 4) LSTM layer 0 gates: rnn_in @ W_ih0^T + h0[0] @ W_hh0^T + b_ih0 + b_hh0
    gemm(rnn_in, XDIM, W_ih0, XDIM, gates0, 4 * H, B, 4 * H, XDIM, 0, b_ih0, b_hh0);
    gemm(h0, H, W_hh0, H, gates0, 4 * H, B, 4 * H, H, 1, nullptr, nullptr);
    hipLaunchKernelGGL(lstm_cell_kernel, dim3((B * H + 255) / 256), dim3(256), 0, stream,
                       gates0, c0, h1, c1);

    // 5) LSTM layer 1 gates: h1 @ W_ih1^T + h0[1] @ W_hh1^T + b_ih1 + b_hh1
    gemm(h1, H, W_ih1, H, gates1, 4 * H, B, 4 * H, H, 0, b_ih1, b_hh1);
    gemm(h0 + (size_t)B * H, H, W_hh1, H, gates1, 4 * H, B, 4 * H, H, 1, nullptr, nullptr);
    hipLaunchKernelGGL(lstm_cell_kernel, dim3((B * H + 255) / 256), dim3(256), 0, stream,
                       gates1, c0 + (size_t)B * H, h2, c2);

    // 6) logits = h2 @ W_out^T + b_out, straight into d_out's final rows
    gemm(h2, H, W_out, H, finalp, VS, B, V, H, 0, b_out, nullptr);

    // 7) p_gen
    hipLaunchKernelGGL(pgen_kernel, dim3((B + 3) / 4), dim3(128), 0, stream,
                       h2, rnn_in, emb, W_pg, b_pg, pgen);

    // 8) softmax * p_gen + 1e-10 (and tail), scatter (1-p_gen)*attn, then log
    hipLaunchKernelGGL(softmax_out_kernel, dim3(B), dim3(1024), 0, stream, finalp, pgen);
    hipLaunchKernelGGL(scatter_kernel, dim3((B * S + 255) / 256), dim3(256), 0, stream,
                       finalp, extv, attn, pgen);
    hipLaunchKernelGGL(log_kernel, dim3((B * VS + 255) / 256), dim3(256), 0, stream,
                       finalp, B * VS);
}
